// AttentionLateralOp_1477468749878
// MI455X (gfx1250) — matrix operand order repaired and checked by reading, not compiled, not run on hardware
//
#include <hip/hip_runtime.h>

#define BB 4
#define CC 512
#define NN 4096
#define CQ 64

typedef __attribute__((ext_vector_type(16))) __bf16 v16bf;
typedef __attribute__((ext_vector_type(2)))  __bf16 v2bf;
typedef __attribute__((ext_vector_type(8)))  float  v8f;
typedef __attribute__((ext_vector_type(4)))  unsigned int v4u;
typedef __attribute__((ext_vector_type(8)))  unsigned int v8u;

union BF16Frag {
    v16bf v;
    unsigned short u[16];
    unsigned int   d[8];
};

// round-to-nearest-even f32 -> bf16 (inputs are well-behaved; no NaN path)
__device__ __forceinline__ unsigned short f2bf_bits(float x) {
    unsigned u = __builtin_bit_cast(unsigned, x);
    return (unsigned short)((u + 0x7FFFu + ((u >> 16) & 1u)) >> 16);
}
__device__ __forceinline__ unsigned int pack_bf16(float lo, float hi) {
#if __has_builtin(__builtin_amdgcn_cvt_pk_bf16_f32)
    v2bf p = __builtin_amdgcn_cvt_pk_bf16_f32(lo, hi);
    return __builtin_bit_cast(unsigned int, p);
#else
    return (unsigned)f2bf_bits(lo) | ((unsigned)f2bf_bits(hi) << 16);
#endif
}

// 16-bit A-matrix (16x32) K base per ISA 7.12.2: lanes 0-15 cover K {0..7,16..23},
// lanes 16-31 cover K {8..15,24..31}; VGPR r holds packed pair (K,K+1).
__device__ __forceinline__ int kbaseA(int r, int h) {
    return ((r < 4) ? 2 * r : 2 * r + 8) + 8 * h;
}

// exchange with lane^16 via ds_swizzle group-of-32 (xor=0x10, and=0x1f)
__device__ __forceinline__ float swap16(float x) {
    int i = __builtin_bit_cast(int, x);
    i = __builtin_amdgcn_ds_swizzle(i, 0x401F);
    return __builtin_bit_cast(float, i);
}

__device__ __forceinline__ v8f wmma_bf16(const BF16Frag& a, const BF16Frag& b, v8f c) {
    return __builtin_amdgcn_wmma_f32_16x16x32_bf16(false, a.v, false, b.v,
                                                   (short)0, c, false, false);
}

// ---------------------------------------------------------------------------
// Elementwise f32 -> bf16 (weights, converted exactly once)
// ---------------------------------------------------------------------------
__global__ void cvt_bf16_kernel(const float* __restrict__ src,
                                unsigned short* __restrict__ dst, int n) {
    const int i = blockIdx.x * blockDim.x + threadIdx.x;
    if (i < n) dst[i] = f2bf_bits(src[i]);
}

// ---------------------------------------------------------------------------
// f/g projection: Y^T[n][64] = (W[64x512] * X[512x4096])^T, bf16 out.
// Block: (b, 128 columns). X tile [32k x 128n] staged f32->bf16 into LDS once
// per k-step by all 256 threads; wave w consumes columns 16w..16w+15 and
// covers all 64 output rows (4 WMMA accumulators).
// ---------------------------------------------------------------------------
__global__ void proj_fg_kernel(const unsigned short* __restrict__ Wbf,  // [64][512] bf16
                               const float* __restrict__ X,             // [B][512][4096]
                               unsigned short* __restrict__ YT) {       // [B][4096][64]
    __shared__ unsigned short ldsX[128 * 32];       // [n][k], k contiguous, 8 KB
    const int lane = threadIdx.x & 31;
    const int wave = threadIdx.x >> 5;
    const int l15 = lane & 15, half = lane >> 4;
    const int b  = blockIdx.x / (NN / 128);
    const int n0 = (blockIdx.x % (NN / 128)) * 128;
    const float* Xb = X + (size_t)b * CC * NN;
    unsigned short* Yb = YT + (size_t)b * NN * CQ;

    v8f zero = {};
    v8f acc[4];
    #pragma unroll
    for (int t = 0; t < 4; ++t) acc[t] = zero;

    for (int kk = 0; kk < CC; kk += 32) {
        #pragma unroll
        for (int i = 0; i < 8; ++i) {               // 2048 bf16 pairs, 8 per thread
            const int p = threadIdx.x + 256 * i;
            const int n = p & 127, k = (p >> 7) * 2;
            const float x0 = Xb[(size_t)(kk + k)     * NN + n0 + n];
            const float x1 = Xb[(size_t)(kk + k + 1) * NN + n0 + n];
            *(unsigned int*)&ldsX[n * 32 + k] = pack_bf16(x0, x1);
        }
        __syncthreads();
        BF16Frag bfr;
        #pragma unroll
        for (int r = 0; r < 8; ++r)
            bfr.d[r] = *(const unsigned int*)&ldsX[(16 * wave + l15) * 32 + kbaseA(r, half)];
        #pragma unroll
        for (int t = 0; t < 4; ++t) {
            BF16Frag a;
            #pragma unroll
            for (int r = 0; r < 8; ++r)
                a.d[r] = *(const unsigned int*)(Wbf + (size_t)(16 * t + l15) * CC + kk + kbaseA(r, half));
            acc[t] = wmma_bf16(a, bfr, acc[t]);
        }
        __syncthreads();
    }
    // rows m = 16t + 8*half + 0..7 are contiguous per lane -> packed dword stores
    #pragma unroll
    for (int t = 0; t < 4; ++t)
        #pragma unroll
        for (int i = 0; i < 4; ++i)
            *(unsigned int*)&Yb[(size_t)(n0 + 16 * wave + l15) * CQ + 16 * t + 8 * half + 2 * i] =
                pack_bf16(acc[t][2 * i], acc[t][2 * i + 1]);
}

// ---------------------------------------------------------------------------
// hh projection: hh[512][4096] = Wv[512x512] * X[512x4096], bf16 row-major.
// Block: (b, 16 columns). X tile [32k x 16n] staged into LDS once per k-step
// (one bf16 pair per thread); all 8 waves share it as the B fragment, wave w
// produces rows 64w..64w+63.
// ---------------------------------------------------------------------------
__global__ void proj_hh_kernel(const unsigned short* __restrict__ Wbf,  // [512][512] bf16
                               const float* __restrict__ X,
                               unsigned short* __restrict__ Y) {        // [B][512][4096]
    __shared__ unsigned short ldsX[16 * 32];        // [n][k], 1 KB
    const int lane = threadIdx.x & 31;
    const int wave = threadIdx.x >> 5;
    const int l15 = lane & 15, half = lane >> 4;
    const int b  = blockIdx.x / (NN / 16);
    const int n0 = (blockIdx.x % (NN / 16)) * 16;
    const float* Xb = X + (size_t)b * CC * NN;
    unsigned short* Yb = Y + (size_t)b * CC * NN;
    const int c0 = wave * 64;

    v8f zero = {};
    v8f acc[4];
    #pragma unroll
    for (int t = 0; t < 4; ++t) acc[t] = zero;

    for (int kk = 0; kk < CC; kk += 32) {
        {                                           // 256 pairs, 1 per thread
            const int n = threadIdx.x & 15, k = (threadIdx.x >> 4) * 2;
            const float x0 = Xb[(size_t)(kk + k)     * NN + n0 + n];
            const float x1 = Xb[(size_t)(kk + k + 1) * NN + n0 + n];
            *(unsigned int*)&ldsX[n * 32 + k] = pack_bf16(x0, x1);
        }
        __syncthreads();
        BF16Frag bfr;
        #pragma unroll
        for (int r = 0; r < 8; ++r)
            bfr.d[r] = *(const unsigned int*)&ldsX[l15 * 32 + kbaseA(r, half)];
        #pragma unroll
        for (int t = 0; t < 4; ++t) {
            BF16Frag a;
            #pragma unroll
            for (int r = 0; r < 8; ++r)
                a.d[r] = *(const unsigned int*)(Wbf + (size_t)(c0 + 16 * t + l15) * CC + kk + kbaseA(r, half));
            acc[t] = wmma_bf16(a, bfr, acc[t]);
        }
        __syncthreads();
    }
    #pragma unroll
    for (int t = 0; t < 4; ++t)
        #pragma unroll
        for (int r = 0; r < 8; ++r)
            Yb[(size_t)(c0 + 16 * t + r + 8 * half) * NN + n0 + l15] = f2bf_bits(acc[t][r]);
}

// ---------------------------------------------------------------------------
// Pass 2: online softmax stats (per-column running max / sum-of-exp),
// scores recomputed with bf16 WMMA and discarded. One wave per (b, 16 cols).
// ---------------------------------------------------------------------------
__global__ void stats_kernel(const unsigned short* __restrict__ fT,
                             const unsigned short* __restrict__ gT,
                             float* __restrict__ colmax, float* __restrict__ colsum) {
    const int lane = threadIdx.x & 31;
    const int l15 = lane & 15, half = lane >> 4;
    const int b  = blockIdx.x / (NN / 16);
    const int n0 = (blockIdx.x % (NN / 16)) * 16;
    const unsigned short* fTb = fT + (size_t)b * NN * CQ;
    const unsigned short* gTb = gT + (size_t)b * NN * CQ;

    BF16Frag gb[2];
    #pragma unroll
    for (int ks = 0; ks < 2; ++ks)
        #pragma unroll
        for (int r = 0; r < 8; ++r)
            gb[ks].d[r] = *(const unsigned int*)(gTb + (size_t)(n0 + l15) * CQ
                                                 + 32 * ks + kbaseA(r, half));

    float runmax = -1e30f, runsum = 0.0f;
    v8f zero = {};

    for (int mt = 0; mt < NN / 32; ++mt) {
        const int m0 = mt * 32;
        v8f s0 = zero, s1 = zero;
        #pragma unroll
        for (int ks = 0; ks < 2; ++ks) {
            BF16Frag a0, a1;
            #pragma unroll
            for (int r = 0; r < 8; ++r) {
                const int K = 32 * ks + kbaseA(r, half);
                a0.d[r] = *(const unsigned int*)(fTb + (size_t)(m0 + l15) * CQ + K);
                a1.d[r] = *(const unsigned int*)(fTb + (size_t)(m0 + 16 + l15) * CQ + K);
            }
            s0 = wmma_bf16(a0, gb[ks], s0);
            s1 = wmma_bf16(a1, gb[ks], s1);
        }
        float tm = -1e30f;
        #pragma unroll
        for (int r = 0; r < 8; ++r) tm = fmaxf(tm, fmaxf(s0[r], s1[r]));
        tm = fmaxf(tm, swap16(tm));
        const float nm = fmaxf(runmax, tm);
        float ts = 0.0f;
        #pragma unroll
        for (int r = 0; r < 8; ++r) ts += __expf(s0[r] - nm) + __expf(s1[r] - nm);
        ts += swap16(ts);
        runsum = runsum * __expf(runmax - nm) + ts;
        runmax = nm;
    }
    if (lane < 16) {
        colmax[(size_t)b * NN + n0 + lane] = runmax;
        colsum[(size_t)b * NN + n0 + lane] = runsum;
    }
}

// ---------------------------------------------------------------------------
// Pass 3: o = gamma * (hh @ beta) + x_t. One workgroup (8 waves) per
// (b, 16 cols). Double-buffered TDM pipeline: wave 7 issues the
// TENSOR_LOAD_TO_LDS DMA for the *next* hh tile ([512 x 32] bf16, 32 KB) and
// waits tensorcnt<=1 so the current tile's DMA has landed while the next one
// flies; wave 0 concurrently recomputes the score tile and writes softmaxed
// beta to LDS; after the barrier all 8 waves run the hh@beta WMMAs from LDS.
// ---------------------------------------------------------------------------
__global__ void out_kernel(const unsigned short* __restrict__ fT,
                           const unsigned short* __restrict__ gT,
                           const unsigned short* __restrict__ hh,
                           const float* __restrict__ colmax,
                           const float* __restrict__ colsum,
                           const float* __restrict__ xt,
                           const float* __restrict__ gamma,
                           float* __restrict__ out) {
    __shared__ unsigned short ldsHH[2][CC * 32];    // [c][m], m contiguous, 2x32 KB
    __shared__ unsigned short ldsBeta[16 * 32];     // [n][m], m contiguous, 1 KB
    const int lane = threadIdx.x & 31;
    const int wave = threadIdx.x >> 5;
    const int l15 = lane & 15, half = lane >> 4;
    const int b  = blockIdx.x / (NN / 16);
    const int n0 = (blockIdx.x % (NN / 16)) * 16;
    const unsigned short* fTb = fT + (size_t)b * NN * CQ;
    const unsigned short* gTb = gT + (size_t)b * NN * CQ;
    const unsigned short* hhb = hh + (size_t)b * CC * NN;
    const int c0 = wave * 64;
    const int MT = NN / 32;

    // D# group1 (loop-invariant): data_size=2B; tensor 4096x512, stride 4096;
    // tile 32x512 (rows = all 512 channels, 32 m-columns)
    const v8u g1 = { 0x00010000u,
                     (unsigned)((NN & 0xFFFF) << 16),
                     (unsigned)((NN >> 16) | ((CC & 0xFFFF) << 16)),
                     (unsigned)((CC >> 16) | (32u << 16)),
                     (unsigned)CC,
                     (unsigned)NN,
                     0u, 0u };
    const unsigned ldsAddr0 = (unsigned)(size_t)&ldsHH[0][0];
    const unsigned ldsAddr1 = (unsigned)(size_t)&ldsHH[1][0];

    BF16Frag gb[2];
    float cm = 0.0f, inv = 0.0f;
    if (wave == 0) {
        #pragma unroll
        for (int ks = 0; ks < 2; ++ks)
            #pragma unroll
            for (int r = 0; r < 8; ++r)
                gb[ks].d[r] = *(const unsigned int*)(gTb + (size_t)(n0 + l15) * CQ
                                                     + 32 * ks + kbaseA(r, half));
        cm  = colmax[(size_t)b * NN + n0 + l15];
        inv = 1.0f / colsum[(size_t)b * NN + n0 + l15];
    }
    if (wave == 7) {                                // prologue: DMA tile 0 -> buf 0
        const unsigned long long ga = (unsigned long long)(size_t)hhb;
        const v4u g0 = { 1u, ldsAddr0, (unsigned)ga,
                         (unsigned)((ga >> 32) & 0x01FFFFFFu) | 0x80000000u };
        asm volatile("tensor_load_to_lds %0, %1" :: "s"(g0), "s"(g1) : "memory");
    }

    v8f zero = {};
    v8f acc[4];
    #pragma unroll
    for (int t = 0; t < 4; ++t) acc[t] = zero;

    for (int mt = 0; mt < MT; ++mt) {
        const int m0 = mt * 32;
        if (wave == 7) {
            if (mt + 1 < MT) {                      // issue next tile into other buffer
                const unsigned long long ga =
                    (unsigned long long)(size_t)(hhb + m0 + 32);
                const v4u g0 = { 1u, (mt & 1) ? ldsAddr0 : ldsAddr1, (unsigned)ga,
                                 (unsigned)((ga >> 32) & 0x01FFFFFFu) | 0x80000000u };
                asm volatile("tensor_load_to_lds %0, %1" :: "s"(g0), "s"(g1) : "memory");
                __builtin_amdgcn_s_wait_tensorcnt(1);   // current tile landed
            } else {
                __builtin_amdgcn_s_wait_tensorcnt(0);
            }
        }
        if (wave == 0) {
            v8f s0 = zero, s1 = zero;
            #pragma unroll
            for (int ks = 0; ks < 2; ++ks) {
                BF16Frag a0, a1;
                #pragma unroll
                for (int r = 0; r < 8; ++r) {
                    const int K = 32 * ks + kbaseA(r, half);
                    a0.d[r] = *(const unsigned int*)(fTb + (size_t)(m0 + l15) * CQ + K);
                    a1.d[r] = *(const unsigned int*)(fTb + (size_t)(m0 + 16 + l15) * CQ + K);
                }
                s0 = wmma_bf16(a0, gb[ks], s0);
                s1 = wmma_bf16(a1, gb[ks], s1);
            }
            // rows 8*half..8*half+7 (and +16) are contiguous -> packed stores
            #pragma unroll
            for (int i = 0; i < 4; ++i) {
                *(unsigned int*)&ldsBeta[l15 * 32 + 8 * half + 2 * i] =
                    pack_bf16(__expf(s0[2 * i] - cm) * inv, __expf(s0[2 * i + 1] - cm) * inv);
                *(unsigned int*)&ldsBeta[l15 * 32 + 16 + 8 * half + 2 * i] =
                    pack_bf16(__expf(s1[2 * i] - cm) * inv, __expf(s1[2 * i + 1] - cm) * inv);
            }
            if (mt + 1 < MT)                        // global_prefetch_b8 for next f tile
                __builtin_prefetch(fTb + (size_t)(m0 + 32 + l15) * CQ, 0, 1);
        }
        __syncthreads();

        const unsigned short* hhT = &ldsHH[mt & 1][0];
        BF16Frag betaF;
        #pragma unroll
        for (int r = 0; r < 8; ++r)
            betaF.d[r] = *(const unsigned int*)&ldsBeta[l15 * 32 + kbaseA(r, half)];
        #pragma unroll
        for (int t = 0; t < 4; ++t) {
            BF16Frag aF;
            #pragma unroll
            for (int r = 0; r < 8; ++r)
                aF.d[r] = *(const unsigned int*)&hhT[(c0 + 16 * t + l15) * 32 + kbaseA(r, half)];
            acc[t] = wmma_bf16(aF, betaF, acc[t]);
        }
        __syncthreads();
    }

    const float gm = gamma[0];
    #pragma unroll
    for (int t = 0; t < 4; ++t)
        #pragma unroll
        for (int r = 0; r < 8; ++r) {
            const int row = c0 + 16 * t + r + 8 * half;
            const size_t idx = ((size_t)b * CC + row) * NN + n0 + l15;
            out[idx] = gm * acc[t][r] + xt[idx];
        }
}

extern "C" void kernel_launch(void* const* d_in, const int* in_sizes, int n_in,
                              void* d_out, int out_size, void* d_ws, size_t ws_size,
                              hipStream_t stream) {
    (void)in_sizes; (void)n_in; (void)out_size; (void)ws_size;
    const float* origin = (const float*)d_in[0];
    const float* target = (const float*)d_in[1];
    const float* Wq     = (const float*)d_in[2];
    const float* Wk     = (const float*)d_in[3];
    const float* Wv     = (const float*)d_in[4];
    const float* gamma  = (const float*)d_in[5];
    float* out = (float*)d_out;

    // workspace: fT/gT [B][N][CQ] bf16, hh [B][C][N] bf16, colmax/colsum [B][N] f32,
    //            bf16 copies of Wq/Wk/Wv  (~20.8 MB total)
    unsigned short* fT = (unsigned short*)d_ws;
    unsigned short* gT = fT + (size_t)BB * NN * CQ;
    unsigned short* hh = gT + (size_t)BB * NN * CQ;
    float* colmax = (float*)(hh + (size_t)BB * CC * NN);
    float* colsum = colmax + (size_t)BB * NN;
    unsigned short* Wqbf = (unsigned short*)(colsum + (size_t)BB * NN);
    unsigned short* Wkbf = Wqbf + (size_t)CQ * CC;
    unsigned short* Wvbf = Wkbf + (size_t)CQ * CC;

    const int nWqk = CQ * CC, nWv = CC * CC;
    cvt_bf16_kernel<<<(nWqk + 255) / 256, 256, 0, stream>>>(Wq, Wqbf, nWqk);
    cvt_bf16_kernel<<<(nWqk + 255) / 256, 256, 0, stream>>>(Wk, Wkbf, nWqk);
    cvt_bf16_kernel<<<(nWv  + 255) / 256, 256, 0, stream>>>(Wv, Wvbf, nWv);

    proj_fg_kernel<<<BB * (NN / 128), 256, 0, stream>>>(Wqbf, target, fT);
    proj_fg_kernel<<<BB * (NN / 128), 256, 0, stream>>>(Wkbf, origin, gT);
    proj_hh_kernel<<<BB * (NN / 16),  256, 0, stream>>>(Wvbf, origin, hh);

    stats_kernel<<<BB * (NN / 16), 32, 0, stream>>>(fT, gT, colmax, colsum);
    out_kernel<<<BB * (NN / 16), 256, 0, stream>>>(fT, gT, hh, colmax, colsum,
                                                   target, gamma, out);
}
